// GPTBlock_25099788877978
// MI455X (gfx1250) — compile-verified, matrix operand order checked by reading, not run on hardware
//
#include <hip/hip_runtime.h>
#include <hip/hip_bf16.h>

// ---------------------------------------------------------------------------
// GPT block for gfx1250 (MI455X): pre-LN MHA (causal flash attention) + FFN.
// - All matmuls via v_wmma_f32_16x16x32_f16 (wave32).
// - GEMM operand staging via the Tensor Data Mover (tensor_load_to_lds),
//   double-buffered LDS, TENSORcnt-pipelined.
// - Weights cast to f16 once per launch; fc1/fc2 pre-transposed so every
//   GEMM is NT-form (K contiguous for both operands -> shared frag loader).
// ---------------------------------------------------------------------------

typedef __attribute__((ext_vector_type(16))) _Float16     v16h;
typedef __attribute__((ext_vector_type(8)))  _Float16     v8h;
typedef __attribute__((ext_vector_type(8)))  float        v8f;
typedef __attribute__((ext_vector_type(4)))  unsigned int u32x4;
typedef __attribute__((ext_vector_type(8)))  int          i32x8;
typedef __attribute__((ext_vector_type(4)))  int          i32x4;

constexpr int kB   = 2;
constexpr int kL   = 2048;
constexpr int kD   = 1024;
constexpr int kH   = 16;
constexpr int kDH  = 64;
constexpr int kDFF = 4096;
constexpr int kBL  = kB * kL;
constexpr float kEPS = 1e-5f;

// ---------------------------------------------------------------------------
// WMMA helpers
// ---------------------------------------------------------------------------
__device__ __forceinline__ v8f wmma_f16(v16h a, v16h b, v8f c) {
  return __builtin_amdgcn_wmma_f32_16x16x32_f16(
      false, a, false, b, (short)0, c, false, false);
}

// NT fragment loader: operand row-major, K contiguous (ld = row stride in
// elements). Matches the ISA 16-bit A/B striping:
//   lanes 0-15 : row=lane,    K = {0..7, 16..23}
//   lanes 16-31: row=lane-16, K = {8..15, 24..31}
__device__ __forceinline__ v16h load_frag(const _Float16* __restrict__ p,
                                          int ld, int lane) {
  int r  = lane & 15;
  int k0 = (lane >> 4) << 3;           // 0 or 8
  const _Float16* q = p + r * ld + k0;
  union { v16h v; v8h h[2]; } u;
  u.h[0] = *(const v8h*)(q);
  u.h[1] = *(const v8h*)(q + 16);
  return u.v;
}

// ---------------------------------------------------------------------------
// Tensor Data Mover: 2-D tile load Global -> LDS.
// D# per CDNA5 ISA ch.8: group0 = {flags, lds_addr, global_addr, type=2},
// group1 = {data_size=1(2B), tensor dims, tile dims, dim0 stride}.
// Issued by one wave (EXEC ignored by TDM); tracked with TENSORcnt.
// ---------------------------------------------------------------------------
__device__ __forceinline__ void tdm_load_2d(unsigned lds_off,
                                            const _Float16* gptr,
                                            unsigned tile_k,    // elements
                                            unsigned tile_rows, // rows
                                            unsigned long long row_stride) {
  unsigned long long ga = (unsigned long long)(__UINTPTR_TYPE__)gptr;
  u32x4 g0;
  g0[0] = 1u;                                     // count=1, user descriptor
  g0[1] = lds_off;                                // lds_addr (bytes)
  g0[2] = (unsigned)(ga & 0xffffffffull);         // global_addr[31:0]
  g0[3] = (unsigned)((ga >> 32) & 0x1ffffffull)   // global_addr[56:32]
          | (2u << 30);                           // type = 2 ("image")
  i32x8 g1;
  g1[0] = 0x00010000;                             // data_size=1 (2 bytes)
  g1[1] = (int)((tile_k & 0xffffu) << 16);        // tensor_dim0[15:0]
  g1[2] = (int)(((tile_k >> 16) & 0xffffu)        // tensor_dim0[31:16]
          | ((tile_rows & 0xffffu) << 16));       // tensor_dim1[15:0]
  g1[3] = (int)(((tile_rows >> 16) & 0xffffu)     // tensor_dim1[31:16]
          | ((tile_k & 0xffffu) << 16));          // tile_dim0
  g1[4] = (int)(tile_rows & 0xffffu);             // tile_dim1 (tile_dim2=0)
  g1[5] = (int)(row_stride & 0xffffffffull);      // tensor_dim0_stride[31:0]
  g1[6] = (int)((row_stride >> 32) & 0xffffull);  // stride[47:32]
  g1[7] = 0;
  i32x4 z4 = {};
#if defined(__clang_major__) && __clang_major__ >= 23
  i32x8 z8 = {};
  __builtin_amdgcn_tensor_load_to_lds(g0, g1, z4, z4, z8, 0);
#else
  __builtin_amdgcn_tensor_load_to_lds(g0, g1, z4, z4, 0);
#endif
}

__device__ __forceinline__ unsigned lds_offset_of(const void* p) {
  // __shared__ generic pointers carry the LDS byte offset in the low 32 bits
  // (LDS aperture mapping discards the upper bits).
  return (unsigned)(__UINTPTR_TYPE__)p;
}

// ---------------------------------------------------------------------------
// Weight conversion kernels (fp32 -> f16, optionally transposed)
// ---------------------------------------------------------------------------
__global__ void cvt_f16(const float* __restrict__ in, _Float16* __restrict__ out,
                        int n) {
  for (int i = blockIdx.x * blockDim.x + threadIdx.x; i < n;
       i += gridDim.x * blockDim.x)
    out[i] = (_Float16)in[i];
}

// in[rows][cols] -> out[cols][rows]
__global__ void cvt_f16_t(const float* __restrict__ in, _Float16* __restrict__ out,
                          int rows, int cols) {
  int n = rows * cols;
  for (int i = blockIdx.x * blockDim.x + threadIdx.x; i < n;
       i += gridDim.x * blockDim.x) {
    int r = i / cols, c = i - r * cols;
    out[c * rows + r] = (_Float16)in[i];
  }
}

// ---------------------------------------------------------------------------
// LayerNorm: one block per token (D=1024), 256 threads x 4 elements
// ---------------------------------------------------------------------------
__global__ __launch_bounds__(256)
void layernorm_f16(const float* __restrict__ x, const float* __restrict__ gamma,
                   const float* __restrict__ beta, _Float16* __restrict__ out) {
  int row  = blockIdx.x;
  int tid  = threadIdx.x;
  int lane = tid & 31;
  int wave = tid >> 5;
  const float4 v = ((const float4*)(x + (size_t)row * kD))[tid];

  float s  = v.x + v.y + v.z + v.w;
  float ss = v.x * v.x + v.y * v.y + v.z * v.z + v.w * v.w;
  for (int off = 16; off; off >>= 1) {
    s  += __shfl_down(s, off);
    ss += __shfl_down(ss, off);
  }
  __shared__ float red[2][8];
  if (lane == 0) { red[0][wave] = s; red[1][wave] = ss; }
  __syncthreads();
  float S = 0.f, SS = 0.f;
#pragma unroll
  for (int w = 0; w < 8; ++w) { S += red[0][w]; SS += red[1][w]; }
  float mu   = S * (1.0f / kD);
  float var  = SS * (1.0f / kD) - mu * mu;
  float rinv = rsqrtf(var + kEPS);

  int c = tid * 4;
  _Float16* o = out + (size_t)row * kD + c;
  o[0] = (_Float16)((v.x - mu) * rinv * gamma[c + 0] + beta[c + 0]);
  o[1] = (_Float16)((v.y - mu) * rinv * gamma[c + 1] + beta[c + 1]);
  o[2] = (_Float16)((v.z - mu) * rinv * gamma[c + 2] + beta[c + 2]);
  o[3] = (_Float16)((v.w - mu) * rinv * gamma[c + 3] + beta[c + 3]);
}

// ---------------------------------------------------------------------------
// NT GEMM, TDM-staged, double-buffered LDS.
//   C[m,n] = sum_k A[m,k] * Bw[n,k]     (A: MxK f16, Bw: NxK f16 row-major)
// Block tile 256x128 (8 waves, 4x2), wave tile 64x64 = 4x4 WMMA, BK=32.
// Wave 0 drives the TDM; s_wait_tensorcnt(2) overlaps DMA with compute.
// ---------------------------------------------------------------------------
enum { EPI_HEAD = 0, EPI_VT = 1, EPI_O = 2, EPI_FC1 = 3, EPI_FC2 = 4 };
constexpr int BM = 256, BN = 128, BK = 32;

template <int EPI>
__global__ __launch_bounds__(256)
void gemm_nt(const _Float16* __restrict__ A, const _Float16* __restrict__ Bw,
             const float* __restrict__ bias, const float* __restrict__ resid,
             float* __restrict__ outF, _Float16* __restrict__ outH,
             int M, int N, int K, float scale) {
  __shared__ __align__(16) _Float16 As[2][BM * BK];   // 2 x 16 KB
  __shared__ __align__(16) _Float16 Bs[2][BN * BK];   // 2 x  8 KB

  const int tid  = threadIdx.x;
  const int lane = tid & 31;
  const int wave = tid >> 5;
  const int wm   = wave & 3;    // 0..3 -> 64-row stripe
  const int wn   = wave >> 2;   // 0..1 -> 64-col stripe
  const int m0   = blockIdx.x * BM;
  const int n0   = blockIdx.y * BN;

  const _Float16* Ag = A  + (size_t)m0 * K;
  const _Float16* Bg = Bw + (size_t)n0 * K;

  v8f acc[4][4] = {};

  if (wave == 0) {   // prologue: fill buffer 0
    tdm_load_2d(lds_offset_of(&As[0][0]), Ag, BK, BM, (unsigned long long)K);
    tdm_load_2d(lds_offset_of(&Bs[0][0]), Bg, BK, BN, (unsigned long long)K);
  }

  int buf = 0;
  for (int k0 = 0; k0 < K; k0 += BK) {
    if (wave == 0) {
      if (k0 + BK < K) {   // prefetch next stage into the other buffer
        tdm_load_2d(lds_offset_of(&As[buf ^ 1][0]), Ag + k0 + BK, BK, BM,
                    (unsigned long long)K);
        tdm_load_2d(lds_offset_of(&Bs[buf ^ 1][0]), Bg + k0 + BK, BK, BN,
                    (unsigned long long)K);
        __builtin_amdgcn_s_wait_tensorcnt(2);  // current stage complete
      } else {
        __builtin_amdgcn_s_wait_tensorcnt(0);
      }
    }
    __syncthreads();

    v16h a0 = load_frag(&As[buf][(wm * 64 + 0)  * BK], BK, lane);
    v16h a1 = load_frag(&As[buf][(wm * 64 + 16) * BK], BK, lane);
    v16h a2 = load_frag(&As[buf][(wm * 64 + 32) * BK], BK, lane);
    v16h a3 = load_frag(&As[buf][(wm * 64 + 48) * BK], BK, lane);
#pragma unroll
    for (int j = 0; j < 4; ++j) {
      v16h b = load_frag(&Bs[buf][(wn * 64 + j * 16) * BK], BK, lane);
      acc[0][j] = wmma_f16(a0, b, acc[0][j]);
      acc[1][j] = wmma_f16(a1, b, acc[1][j]);
      acc[2][j] = wmma_f16(a2, b, acc[2][j]);
      acc[3][j] = wmma_f16(a3, b, acc[3][j]);
    }
    __syncthreads();   // stage consumed; TDM may overwrite it next iteration
    buf ^= 1;
  }

  // Epilogue. C-tile layout: lanes 0-15 rows 0..7 (vgpr r), lanes 16-31 rows 8..15.
  const int nn = lane & 15;
  const int mo = (lane >> 4) * 8;
#pragma unroll
  for (int i = 0; i < 4; ++i) {
#pragma unroll
    for (int j = 0; j < 4; ++j) {
      const int mbase = m0 + wm * 64 + i * 16 + mo;
      const int n     = n0 + wn * 64 + j * 16 + nn;
#pragma unroll
      for (int r = 0; r < 8; ++r) {
        const int m = mbase + r;
        float val = acc[i][j][r];
        if constexpr (EPI == EPI_HEAD) {
          // out[b][h][l][dh], f16 (Q gets scale=1/sqrt(DH), K scale=1)
          int b = m >> 11, l = m & (kL - 1), h = n >> 6, dh = n & (kDH - 1);
          outH[(((b * kH + h) * kL) + l) * kDH + dh] = (_Float16)(val * scale);
        } else if constexpr (EPI == EPI_VT) {
          // out[b][h][dh][l] (V transposed for the PV WMMA B-operand)
          int b = m >> 11, l = m & (kL - 1), h = n >> 6, dh = n & (kDH - 1);
          outH[(((b * kH + h) * kDH) + dh) * kL + l] = (_Float16)val;
        } else if constexpr (EPI == EPI_O) {
          outF[(size_t)m * N + n] = resid[(size_t)m * N + n] + val;
        } else if constexpr (EPI == EPI_FC1) {
          float t = val + bias[n];
          outH[(size_t)m * N + n] =
              (_Float16)(0.5f * t * (1.0f + erff(t * 0.70710678118654752f)));
        } else {  // EPI_FC2
          outF[(size_t)m * N + n] = resid[(size_t)m * N + n] + val + bias[n];
        }
      }
    }
  }
}

// ---------------------------------------------------------------------------
// Causal flash attention. Grid: (L/64, B*H), 128 threads = 4 independent waves.
// Wave w owns 16 query rows; DH=64. Q pre-scaled by 1/sqrt(DH).
//   S = Q*K^T (NT WMMA), online softmax, ctx += P*V via Vt (NT WMMA),
//   P re-striped through per-wave LDS into A-fragment layout.
// ---------------------------------------------------------------------------
__global__ __launch_bounds__(128)
void flash_attn(const _Float16* __restrict__ Q, const _Float16* __restrict__ Kh,
                const _Float16* __restrict__ Vt, _Float16* __restrict__ ctxOut) {
  const int bh   = blockIdx.y;
  const int q0   = blockIdx.x * 64;
  const int wave = threadIdx.x >> 5;
  const int lane = threadIdx.x & 31;
  const int qw   = q0 + wave * 16;
  const int nn   = lane & 15;
  const int mo   = (lane >> 4) * 8;

  const _Float16* Qh = Q  + (size_t)bh * kL * kDH + (size_t)qw * kDH;
  const _Float16* Kb = Kh + (size_t)bh * kL * kDH;
  const _Float16* Vb = Vt + (size_t)bh * kDH * kL;

  v16h qf0 = load_frag(Qh + 0,  kDH, lane);
  v16h qf1 = load_frag(Qh + 32, kDH, lane);

  v8f   ctx[4] = {};
  float rmax[8], rsum[8];
#pragma unroll
  for (int r = 0; r < 8; ++r) { rmax[r] = -3.0e38f; rsum[r] = 0.f; }

  __shared__ __align__(16) _Float16 Ps[4][16 * 64];   // per-wave P staging
  _Float16* pw = &Ps[wave][0];

  for (int j0 = 0; j0 <= qw + 15; j0 += 64) {
    // ---- S = Q * K^T for a 16x64 score tile ----
    v8f s[4] = {};
#pragma unroll
    for (int j = 0; j < 4; ++j) {
      v16h kb0 = load_frag(Kb + (size_t)(j0 + j * 16) * kDH + 0,  kDH, lane);
      v16h kb1 = load_frag(Kb + (size_t)(j0 + j * 16) * kDH + 32, kDH, lane);
      s[j] = wmma_f16(qf0, kb0, s[j]);
      s[j] = wmma_f16(qf1, kb1, s[j]);
    }

    // ---- causal mask + online softmax (rows live across 16-lane halves) ----
#pragma unroll
    for (int r = 0; r < 8; ++r) {
      const int qrow = qw + mo + r;
      float mx = -3.0e38f;
#pragma unroll
      for (int j = 0; j < 4; ++j) {
        float sv = ((j0 + j * 16 + nn) > qrow) ? -3.0e38f : s[j][r];
        s[j][r] = sv;
        mx = fmaxf(mx, sv);
      }
      for (int off = 1; off < 16; off <<= 1) mx = fmaxf(mx, __shfl_xor(mx, off));
      const float mnew = fmaxf(rmax[r], mx);
      const float corr = __expf(rmax[r] - mnew);
      rmax[r] = mnew;
      float ps = 0.f;
#pragma unroll
      for (int j = 0; j < 4; ++j) {
        float p = __expf(s[j][r] - mnew);
        s[j][r] = p;
        ps += p;
      }
      for (int off = 1; off < 16; off <<= 1) ps += __shfl_xor(ps, off);
      rsum[r] = rsum[r] * corr + ps;
#pragma unroll
      for (int d = 0; d < 4; ++d) ctx[d][r] *= corr;
    }

    // ---- re-stripe P into A-fragment layout via LDS ----
#pragma unroll
    for (int j = 0; j < 4; ++j)
#pragma unroll
      for (int r = 0; r < 8; ++r)
        pw[(mo + r) * 64 + j * 16 + nn] = (_Float16)s[j][r];

    // ---- ctx += P * V  (B-operand rows = Vt rows, keys contiguous) ----
#pragma unroll
    for (int kk = 0; kk < 2; ++kk) {
      v16h pa = load_frag(pw + kk * 32, 64, lane);
#pragma unroll
      for (int d = 0; d < 4; ++d) {
        v16h vb = load_frag(Vb + (size_t)(d * 16) * kL + j0 + kk * 32, kL, lane);
        ctx[d] = wmma_f16(pa, vb, ctx[d]);
      }
    }
  }

  // ---- normalize + store ctx as [B][L][H][DH] == row-major [B*L, D] ----
  const int b = bh >> 4, h = bh & 15;
#pragma unroll
  for (int r = 0; r < 8; ++r) {
    const float inv = 1.0f / rsum[r];
    const size_t token = (size_t)b * kL + qw + mo + r;
    const int colb = h * kDH + nn;
#pragma unroll
    for (int d = 0; d < 4; ++d)
      ctxOut[token * kD + colb + d * 16] = (_Float16)(ctx[d][r] * inv);
  }
}

// ---------------------------------------------------------------------------
// Host-side orchestration
// ---------------------------------------------------------------------------
extern "C" void kernel_launch(void* const* d_in, const int* in_sizes, int n_in,
                              void* d_out, int out_size, void* d_ws, size_t ws_size,
                              hipStream_t stream) {
  const float* x      = (const float*)d_in[0];
  // d_in[1] = attn_mask (causal, handled analytically in-kernel)
  const float* ln1_g  = (const float*)d_in[2];
  const float* ln1_b  = (const float*)d_in[3];
  const float* W_q    = (const float*)d_in[4];
  const float* W_k    = (const float*)d_in[5];
  const float* W_v    = (const float*)d_in[6];
  const float* W_o    = (const float*)d_in[7];
  const float* ln2_g  = (const float*)d_in[8];
  const float* ln2_b  = (const float*)d_in[9];
  const float* fc1_W  = (const float*)d_in[10];
  const float* fc1_b  = (const float*)d_in[11];
  const float* fc2_W  = (const float*)d_in[12];
  const float* fc2_b  = (const float*)d_in[13];
  float* out = (float*)d_out;

  char* ws = (char*)d_ws;
  auto alloc = [&](size_t bytes) -> char* {
    char* p = ws;
    ws += (bytes + 255) & ~(size_t)255;
    return p;
  };
  _Float16* wq_h  = (_Float16*)alloc((size_t)kD * kD * 2);
  _Float16* wk_h  = (_Float16*)alloc((size_t)kD * kD * 2);
  _Float16* wv_h  = (_Float16*)alloc((size_t)kD * kD * 2);
  _Float16* wo_h  = (_Float16*)alloc((size_t)kD * kD * 2);
  _Float16* fc1t  = (_Float16*)alloc((size_t)kD * kDFF * 2);   // [DFF][D]
  _Float16* fc2t  = (_Float16*)alloc((size_t)kD * kDFF * 2);   // [D][DFF]
  _Float16* hhat  = (_Float16*)alloc((size_t)kBL * kD * 2);
  _Float16* qbuf  = (_Float16*)alloc((size_t)kBL * kD * 2);    // [B][H][L][DH]
  _Float16* kbuf  = (_Float16*)alloc((size_t)kBL * kD * 2);    // [B][H][L][DH]
  _Float16* vtbuf = (_Float16*)alloc((size_t)kBL * kD * 2);    // [B][H][DH][L]
  _Float16* ctxb  = (_Float16*)alloc((size_t)kBL * kD * 2);    // [B*L][D]
  float*    x1    = (float*)   alloc((size_t)kBL * kD * 4);
  _Float16* h2    = (_Float16*)alloc((size_t)kBL * kD * 2);
  _Float16* fbuf  = (_Float16*)alloc((size_t)kBL * kDFF * 2);

  const dim3 blk256(256);
  const dim3 cvtg(2048);

  // Weight precision staging (f16 weights ~24 MB, L2-resident thereafter)
  cvt_f16<<<cvtg, blk256, 0, stream>>>(W_q, wq_h, kD * kD);
  cvt_f16<<<cvtg, blk256, 0, stream>>>(W_k, wk_h, kD * kD);
  cvt_f16<<<cvtg, blk256, 0, stream>>>(W_v, wv_h, kD * kD);
  cvt_f16<<<cvtg, blk256, 0, stream>>>(W_o, wo_h, kD * kD);
  cvt_f16_t<<<cvtg, blk256, 0, stream>>>(fc1_W, fc1t, kD, kDFF);
  cvt_f16_t<<<cvtg, blk256, 0, stream>>>(fc2_W, fc2t, kDFF, kD);

  // LN1 -> h (f16)
  layernorm_f16<<<dim3(kBL), blk256, 0, stream>>>(x, ln1_g, ln1_b, hhat);

  // QKV projections (NT GEMM; Q pre-scaled by 1/sqrt(DH))
  const dim3 gD(kBL / BM, kD / BN);
  gemm_nt<EPI_HEAD><<<gD, blk256, 0, stream>>>(hhat, wq_h, nullptr, nullptr,
                                               nullptr, qbuf, kBL, kD, kD, 0.125f);
  gemm_nt<EPI_HEAD><<<gD, blk256, 0, stream>>>(hhat, wk_h, nullptr, nullptr,
                                               nullptr, kbuf, kBL, kD, kD, 1.0f);
  gemm_nt<EPI_VT><<<gD, blk256, 0, stream>>>(hhat, wv_h, nullptr, nullptr,
                                             nullptr, vtbuf, kBL, kD, kD, 1.0f);

  // Causal flash attention -> ctx [B*L, D]
  flash_attn<<<dim3(kL / 64, kB * kH), dim3(128), 0, stream>>>(qbuf, kbuf,
                                                               vtbuf, ctxb);

  // Output projection + residual 1 -> x1 (fp32)
  gemm_nt<EPI_O><<<gD, blk256, 0, stream>>>(ctxb, wo_h, nullptr, x,
                                            x1, nullptr, kBL, kD, kD, 1.0f);

  // LN2 -> h2 (f16)
  layernorm_f16<<<dim3(kBL), blk256, 0, stream>>>(x1, ln2_g, ln2_b, h2);

  // FFN: fc1 (+bias, exact-erf GELU) -> f16; fc2 (+bias, residual) -> d_out
  const dim3 gF1(kBL / BM, kDFF / BN);
  gemm_nt<EPI_FC1><<<gF1, blk256, 0, stream>>>(h2, fc1t, fc1_b, nullptr,
                                               nullptr, fbuf, kBL, kDFF, kD, 1.0f);
  gemm_nt<EPI_FC2><<<gD, blk256, 0, stream>>>(fbuf, fc2t, fc2_b, x1,
                                              out, nullptr, kBL, kD, kDFF, 1.0f);
  (void)in_sizes; (void)n_in; (void)out_size; (void)ws_size;
}
